// BroadwayGNN_35613868818649
// MI455X (gfx1250) — compile-verified, matrix operand order checked by reading, not run on hardware
//
#include <hip/hip_runtime.h>
#include <stdint.h>

// ---------------------------------------------------------------------------
// BroadwayGNN on MI455X (gfx1250, wave32).
//  - Dense projections: bf16 WMMA (v_wmma_f32_16x16x32_bf16), K=128 ->
//    4 WMMAs per 16x16 tile split over two accumulator chains (ILP across the
//    WMMA->WMMA hazard window), one tile per wave, 8 waves/block.
//  - A strip staged in LDS via CDNA5 async copy (global_load_async_to_lds_b128
//    + s_wait_asynccnt), A fragments then read with ds_load_b128.
//  - COLS is a template parameter so epilogue stores use immediate offsets.
//  - Segment softmax via monotonic-uint atomicMax + f32 atomicAdd (two-pass).
//  - Attention-weighted scatter: one wave per edge, 4 channels/lane, f32
//    atomics into an L2-resident accumulator (25.6 MB << 192 MB L2).
// ---------------------------------------------------------------------------

typedef __bf16 bf16_t;
typedef __attribute__((ext_vector_type(8)))  bf16_t v8bf;
typedef __attribute__((ext_vector_type(16))) bf16_t v16bf;
typedef __attribute__((ext_vector_type(8)))  float  v8f;

#define DEV_INL static __device__ __forceinline__

// ---- CDNA5 async global->LDS copy (CDNA5 ISA) -------------
// VDST VGPR carries the LDS byte offset; a flat pointer into __shared__ holds
// that offset in its low 32 bits (LDS aperture lives in the high word).
DEV_INL void async_load_b128_to_lds(unsigned lds_off, const void* gaddr) {
  asm volatile("global_load_async_to_lds_b128 %0, %1, off"
               :: "v"(lds_off), "v"(gaddr)
               : "memory");
}
DEV_INL void wait_asynccnt0() {
  asm volatile("s_wait_asynccnt 0x0" ::: "memory");
}

// monotonic encoding of float -> uint so atomicMax works for any sign
DEV_INL unsigned f2mono(float f) {
  unsigned u = __float_as_uint(f);
  return (u & 0x80000000u) ? ~u : (u | 0x80000000u);
}
DEV_INL float mono2f(unsigned e) {
  return (e & 0x80000000u) ? __uint_as_float(e ^ 0x80000000u)
                           : __uint_as_float(~e);
}
#define MONO_NEG_INF 0x007FFFFFu  // f2mono(-inf)

// ---------------------------------------------------------------------------
// elementwise conversions
// ---------------------------------------------------------------------------
__global__ void f32_to_bf16_k(const float* __restrict__ x, bf16_t* __restrict__ y, long n) {
  long i = (long)blockIdx.x * blockDim.x + threadIdx.x;
  if (i >= n) return;
  y[i] = (bf16_t)x[i];
}

__global__ void relu_f32_to_bf16_k(const float* __restrict__ x, bf16_t* __restrict__ y, long n) {
  long i = (long)blockIdx.x * blockDim.x + threadIdx.x;
  if (i >= n) return;
  y[i] = (bf16_t)fmaxf(x[i], 0.f);
}

// W is [128 x cols] row-major f32; Wt is [cols x 128] row-major bf16
__global__ void transpose_to_bf_k(const float* __restrict__ W, bf16_t* __restrict__ Wt, int cols) {
  int i = blockIdx.x * blockDim.x + threadIdx.x;
  if (i >= cols * 128) return;
  int n = i >> 7, k = i & 127;
  Wt[i] = (bf16_t)W[(size_t)k * cols + n];
}

// ---------------------------------------------------------------------------
// WMMA GEMM: C[rows x COLS] = X[rows x 128] (bf16) @ Wt^T  (Wt: [COLS x 128])
// rows multiple of 16; COLS in {128, 64}. One 16x16 tile per wave,
// 8 waves/block. The block's A row-strip (16 rows if COLS=128, 32 if COLS=64)
// is staged into LDS once via async copies; fragments per CDNA5 ISA 7.12.2.
// ---------------------------------------------------------------------------
template <int COLS, bool BIAS, bool RELU, bool WF32, bool WBF16>
__global__ __launch_bounds__(256) void gemm128_wmma(
    const bf16_t* __restrict__ X, const bf16_t* __restrict__ Wt,
    const float* __restrict__ bias, float* __restrict__ Cf, bf16_t* __restrict__ Cbf,
    int rows)
{
  constexpr int colTiles    = COLS / 16;             // 8 or 4
  constexpr int stripBytes  = (8 / colTiles) * 16 * 256;  // staged rows * 256 B

  __shared__ __align__(16) bf16_t ldsA[stripBytes / 2];

  const int wave = threadIdx.x >> 5;
  const int lane = threadIdx.x & 31;
  const int totalTiles = (rows >> 4) * colTiles;
  const int tile   = blockIdx.x * 8 + wave;
  const int rtBase = (blockIdx.x * 8) / colTiles;    // first row-tile of block

  // ---- wave 0: stage A strip into LDS with async global->LDS b128 ----
  if (wave == 0) {
    const char* gsrc = (const char*)(X + (size_t)rtBase * 16 * 128);
#pragma unroll
    for (int it0 = 0; it0 < stripBytes; it0 += 32 * 16) {
      const int it = it0 + lane * 16;
      const int grow = rtBase * 16 + (it >> 8);      // 256 B per row
      if (grow < rows) {                             // clamp ragged tail (EXEC-masked)
        async_load_b128_to_lds((unsigned)(uintptr_t)((char*)ldsA + it), gsrc + it);
      }
    }
    wait_asynccnt0();
  }
  __syncthreads();

  if (tile < totalTiles) {                           // wave-uniform: EXEC all-ones inside
    const int rt = tile / colTiles;                  // constexpr divisor -> shifts
    const int ct = tile % colTiles;
    const int r    = lane & 15;
    const int hsel = (lane >> 4) << 3;               // 0 for lanes 0-15, 8 for 16-31

    const bf16_t* Alds = ldsA + (size_t)((rt - rtBase) * 16 + r) * 128;
    const bf16_t* Brow = Wt + (size_t)(ct * 16 + r) * 128;

    // two accumulator chains -> WMMA->WMMA RAW chain length 2 instead of 4
    v8f acc0 = {0.f, 0.f, 0.f, 0.f, 0.f, 0.f, 0.f, 0.f};
    v8f acc1 = {0.f, 0.f, 0.f, 0.f, 0.f, 0.f, 0.f, 0.f};
#pragma unroll
    for (int kc = 0; kc < 4; ++kc) {
      const int kb = kc * 32 + hsel;
      v8bf alo = *(const v8bf*)(Alds + kb);          // ds_load_b128
      v8bf ahi = *(const v8bf*)(Alds + kb + 16);
      v8bf blo = *(const v8bf*)(Brow + kb);          // global_load_b128 (L0/L2 hot)
      v8bf bhi = *(const v8bf*)(Brow + kb + 16);
      v16bf a = __builtin_shufflevector(alo, ahi, 0,1,2,3,4,5,6,7,8,9,10,11,12,13,14,15);
      v16bf b = __builtin_shufflevector(blo, bhi, 0,1,2,3,4,5,6,7,8,9,10,11,12,13,14,15);
      if (kc & 1)
        acc1 = __builtin_amdgcn_wmma_f32_16x16x32_bf16(false, a, false, b, (short)0, acc1,
                                                       false, false);
      else
        acc0 = __builtin_amdgcn_wmma_f32_16x16x32_bf16(false, a, false, b, (short)0, acc0,
                                                       false, false);
    }

    // C layout: N = lane&15 ; VGPR p -> M = p + 8*(lane>=16)
    const int col   = ct * 16 + r;
    const int rbase = rt * 16 + hsel;
    float bc;
    if constexpr (BIAS) bc = bias[col];
    float* __restrict__ cfp  = WF32  ? Cf  + (size_t)rbase * COLS + col : nullptr;
    bf16_t* __restrict__ cbp = WBF16 ? Cbf + (size_t)rbase * COLS + col : nullptr;
#pragma unroll
    for (int p = 0; p < 8; ++p) {
      float v = acc0[p] + acc1[p];
      if constexpr (BIAS) v = v + bc;
      if constexpr (RELU) v = fmaxf(v, 0.f);
      if constexpr (WF32)  cfp[p * COLS] = v;        // immediate-offset stores
      if constexpr (WBF16) cbp[p * COLS] = (bf16_t)v;
    }
  }
}

// ---------------------------------------------------------------------------
// per-row dot products (wave per row)
// ---------------------------------------------------------------------------
__global__ void rowdot128_k(const float* __restrict__ H, const float* __restrict__ v,
                            float* __restrict__ out, int n) {
  int row  = (blockIdx.x * blockDim.x + threadIdx.x) >> 5;
  int lane = threadIdx.x & 31;
  if (row >= n) return;
  float4 h = ((const float4*)(H + (size_t)row * 128))[lane];
  float4 w = ((const float4*)v)[lane];
  float s = h.x * w.x + h.y * w.y + h.z * w.z + h.w * w.w;
#pragma unroll
  for (int o = 16; o > 0; o >>= 1) s += __shfl_xor(s, o, 32);
  if (lane == 0) out[row] = s;
}

__global__ void rowdot64_bias_k(const float* __restrict__ H, const float* __restrict__ v,
                                const float* __restrict__ bias, float* __restrict__ out, int n) {
  int row  = (blockIdx.x * blockDim.x + threadIdx.x) >> 5;
  int lane = threadIdx.x & 31;
  if (row >= n) return;
  float2 h = ((const float2*)(H + (size_t)row * 64))[lane];
  float2 w = ((const float2*)v)[lane];
  float s = h.x * w.x + h.y * w.y;
#pragma unroll
  for (int o = 16; o > 0; o >>= 1) s += __shfl_xor(s, o, 32);
  if (lane == 0) out[row] = s + bias[0];
}

// ---------------------------------------------------------------------------
// segment-softmax + scatter
// ---------------------------------------------------------------------------
__global__ void init_softmax_state_k(unsigned* __restrict__ m, float* __restrict__ denom, int n) {
  int i = blockIdx.x * blockDim.x + threadIdx.x;
  if (i >= n) return;
  m[i] = MONO_NEG_INF;
  denom[i] = 0.f;
}

// acc[i, c] = b1[c] (+ b2[c]) -- HeteroConv 'sum' means both GAT biases add
__global__ void init_acc_bias_k(float* __restrict__ acc, const float* __restrict__ b1,
                                const float* __restrict__ b2, int n) {
  int i = blockIdx.x * blockDim.x + threadIdx.x;
  if (i >= n * 128) return;
  int c = i & 127;
  float v = b1[c];
  if (b2) v += b2[c];
  acc[i] = v;
}

__global__ void edge_logits_max_k(const int* __restrict__ src, const int* __restrict__ dst,
                                  const float* __restrict__ a_s, const float* __restrict__ a_d,
                                  float* __restrict__ ebuf, unsigned* __restrict__ m, int E) {
  int i = blockIdx.x * blockDim.x + threadIdx.x;
  if (i >= E) return;
  float e = a_s[src[i]] + a_d[dst[i]];
  e = (e > 0.f) ? e : 0.2f * e;            // leaky_relu(0.2)
  ebuf[i] = e;
  atomicMax(m + dst[i], f2mono(e));
}

__global__ void edge_exp_sum_k(const int* __restrict__ dst, float* __restrict__ ebuf,
                               const unsigned* __restrict__ m, float* __restrict__ denom, int E) {
  int i = blockIdx.x * blockDim.x + threadIdx.x;
  if (i >= E) return;
  int d = dst[i];
  float ex = __expf(ebuf[i] - mono2f(m[d]));
  ebuf[i] = ex;
  atomicAdd(denom + d, ex);
}

// one wave per edge; 4 channels per lane; atomic f32 accumulation into acc[dst]
__global__ void edge_scatter_k(const int* __restrict__ src, const int* __restrict__ dst,
                               const float* __restrict__ ex, const float* __restrict__ denom,
                               const float* __restrict__ hs, float* __restrict__ acc, int E) {
  int wid  = (blockIdx.x * blockDim.x + threadIdx.x) >> 5;
  int lane = threadIdx.x & 31;
  if (wid >= E) return;
  int s = src[wid], d = dst[wid];
  float alpha = ex[wid] / (denom[d] + 1e-16f);
  float4 h = ((const float4*)(hs + (size_t)s * 128))[lane];
  float* a = acc + (size_t)d * 128 + lane * 4;
  atomicAdd(a + 0, alpha * h.x);
  atomicAdd(a + 1, alpha * h.y);
  atomicAdd(a + 2, alpha * h.z);
  atomicAdd(a + 3, alpha * h.w);
}

// ---------------------------------------------------------------------------
// host orchestration
// ---------------------------------------------------------------------------
static inline int ceil_div(long a, long b) { return (int)((a + b - 1) / b); }

static inline void launch_gemm_plain128(hipStream_t s, const bf16_t* X, const bf16_t* Wt,
                                        float* Cf, int rows) {
  int tiles = (rows / 16) * 8;
  gemm128_wmma<128, false, false, true, false><<<ceil_div(tiles, 8), 256, 0, s>>>(
      X, Wt, nullptr, Cf, nullptr, rows);
}
static inline void launch_gemm_bias_bf16_128(hipStream_t s, const bf16_t* X, const bf16_t* Wt,
                                             const float* bias, bf16_t* Cbf, int rows) {
  int tiles = (rows / 16) * 8;
  gemm128_wmma<128, true, false, false, true><<<ceil_div(tiles, 8), 256, 0, s>>>(
      X, Wt, bias, nullptr, Cbf, rows);
}
static inline void launch_gemm_bias_relu_f32_64(hipStream_t s, const bf16_t* X, const bf16_t* Wt,
                                                const float* bias, float* Cf, int rows) {
  int tiles = (rows / 16) * 4;
  gemm128_wmma<64, true, true, true, false><<<ceil_div(tiles, 8), 256, 0, s>>>(
      X, Wt, bias, Cf, nullptr, rows);
}

struct GatBufs {
  bf16_t* Wt;
  float*  hs;
  float*  hd;
  float*  a_s;
  float*  a_d;
  unsigned* m;
  float*  denom;
  float*  ebuf;
};

static void run_gat(hipStream_t stream,
                    const bf16_t* srcBf, const bf16_t* dstBf, int nSrc, int nDst,
                    const int* ei, int E,
                    const float* Wd, const float* Ws, const float* ad_, const float* as_,
                    float* acc, const GatBufs& B)
{
  const int T = 256;
  // hs = x_src @ Ws ; a_s = hs . as
  transpose_to_bf_k<<<ceil_div(128 * 128, T), T, 0, stream>>>(Ws, B.Wt, 128);
  launch_gemm_plain128(stream, srcBf, B.Wt, B.hs, nSrc);
  rowdot128_k<<<ceil_div(nSrc, 8), T, 0, stream>>>(B.hs, as_, B.a_s, nSrc);
  // hd = x_dst @ Wd ; a_d = hd . ad
  transpose_to_bf_k<<<ceil_div(128 * 128, T), T, 0, stream>>>(Wd, B.Wt, 128);
  launch_gemm_plain128(stream, dstBf, B.Wt, B.hd, nDst);
  rowdot128_k<<<ceil_div(nDst, 8), T, 0, stream>>>(B.hd, ad_, B.a_d, nDst);
  // segment softmax over incoming edges, then weighted scatter
  const int* src = ei;
  const int* dst = ei + E;
  init_softmax_state_k<<<ceil_div(nDst, T), T, 0, stream>>>(B.m, B.denom, nDst);
  edge_logits_max_k<<<ceil_div(E, T), T, 0, stream>>>(src, dst, B.a_s, B.a_d, B.ebuf, B.m, E);
  edge_exp_sum_k<<<ceil_div(E, T), T, 0, stream>>>(dst, B.ebuf, B.m, B.denom, E);
  edge_scatter_k<<<ceil_div((long)E * 32, T), T, 0, stream>>>(src, dst, B.ebuf, B.denom,
                                                             B.hs, acc, E);
}

extern "C" void kernel_launch(void* const* d_in, const int* in_sizes, int n_in,
                              void* d_out, int out_size, void* d_ws, size_t ws_size,
                              hipStream_t stream)
{
  constexpr int NP = 200000, NS = 50000, NT = 2000;
  const int T = 256;

  const float* x_person  = (const float*)d_in[0];
  const float* x_show    = (const float*)d_in[1];
  const float* x_theater = (const float*)d_in[2];
  const int* ei_ai = (const int*)d_in[3];
  const int* ei_pr = (const int*)d_in[4];
  const int* ei_at = (const int*)d_in[5];
  const int E_ai = in_sizes[3] / 2, E_pr = in_sizes[4] / 2, E_at = in_sizes[5] / 2;

  // params flatten order (JAX sorted-dict-key pytree order):
  //   top: 'layers' < 'out1' < 'out2' < 'show_lin'
  //   layer dict: 'acted_in' < 'at' < 'produced'
  //   gat dict  : 'Wd' < 'Ws' < 'ad' < 'as' < 'b'
  auto P = [&](int i) { return (const float*)d_in[i]; };
  // layer l: acted_in @ 6+15l, at @ 6+15l+5, produced @ 6+15l+10
  // head: out1.W=51 out1.b=52 out2.W=53 out2.b=54 show_lin.W=55 show_lin.b=56

  // ---- carve workspace (256B aligned) ----
  char* base = (char*)d_ws;
  size_t off = 0;
  auto carve = [&](size_t bytes) -> char* {
    char* p = base + off;
    off = (off + bytes + 255) & ~(size_t)255;
    return p;
  };
  bf16_t* XpBf  = (bf16_t*)carve((size_t)NP * 128 * 2);
  bf16_t* XsBf  = (bf16_t*)carve((size_t)NS * 128 * 2);
  bf16_t* XtBf  = (bf16_t*)carve((size_t)NT * 128 * 2);
  bf16_t* Wt    = (bf16_t*)carve((size_t)128 * 128 * 2);
  float*  hs    = (float*)carve((size_t)NP * 128 * 4);
  float*  hd    = (float*)carve((size_t)NS * 128 * 4);
  float*  showAcc = (float*)carve((size_t)NS * 128 * 4);
  float*  thrAcc  = (float*)carve((size_t)NT * 128 * 4);
  float*  a_s   = (float*)carve((size_t)NP * 4);
  float*  a_d   = (float*)carve((size_t)NS * 4);
  unsigned* m   = (unsigned*)carve((size_t)NS * 4);
  float*  denom = (float*)carve((size_t)NS * 4);
  float*  ebuf  = (float*)carve((size_t)500000 * 4);
  bf16_t* embBf = (bf16_t*)carve((size_t)NS * 128 * 2);
  float*  hF32  = (float*)carve((size_t)NS * 64 * 4);

  GatBufs B{Wt, hs, hd, a_s, a_d, m, denom, ebuf};

  // bf16 copies of node features
  f32_to_bf16_k<<<ceil_div((long)NP * 128, T), T, 0, stream>>>(x_person, XpBf, (long)NP * 128);
  f32_to_bf16_k<<<ceil_div((long)NS * 128, T), T, 0, stream>>>(x_show, XsBf, (long)NS * 128);
  f32_to_bf16_k<<<ceil_div((long)NT * 128, T), T, 0, stream>>>(x_theater, XtBf, (long)NT * 128);

  for (int l = 0; l < 3; ++l) {
    const int bAI = 6 + 15 * l;       // acted_in
    const int bAT = 6 + 15 * l + 5;   // at
    const int bPR = 6 + 15 * l + 10;  // produced

    // accumulators start at the summed GAT biases
    init_acc_bias_k<<<ceil_div((long)NS * 128, T), T, 0, stream>>>(showAcc, P(bAI + 4),
                                                                   P(bPR + 4), NS);
    init_acc_bias_k<<<ceil_div((long)NT * 128, T), T, 0, stream>>>(thrAcc, P(bAT + 4),
                                                                   nullptr, NT);

    run_gat(stream, XpBf, XsBf, NP, NS, ei_ai, E_ai,
            P(bAI + 0), P(bAI + 1), P(bAI + 2), P(bAI + 3), showAcc, B);
    run_gat(stream, XpBf, XsBf, NP, NS, ei_pr, E_pr,
            P(bPR + 0), P(bPR + 1), P(bPR + 2), P(bPR + 3), showAcc, B);
    run_gat(stream, XsBf, XtBf, NS, NT, ei_at, E_at,
            P(bAT + 0), P(bAT + 1), P(bAT + 2), P(bAT + 3), thrAcc, B);

    // xs = relu(show_acc), xt = relu(thr_acc) -> bf16 for next layer's GEMMs
    relu_f32_to_bf16_k<<<ceil_div((long)NS * 128, T), T, 0, stream>>>(showAcc, XsBf,
                                                                      (long)NS * 128);
    relu_f32_to_bf16_k<<<ceil_div((long)NT * 128, T), T, 0, stream>>>(thrAcc, XtBf,
                                                                      (long)NT * 128);
  }

  // head: emb = xs @ show_lin.W + b  (bf16 out, feeds next GEMM)
  transpose_to_bf_k<<<ceil_div(128 * 128, T), T, 0, stream>>>(P(55), Wt, 128);
  launch_gemm_bias_bf16_128(stream, XsBf, Wt, P(56), embBf, NS);
  // h = relu(emb @ out1.W + out1.b)   [NS x 64]
  transpose_to_bf_k<<<ceil_div(64 * 128, T), T, 0, stream>>>(P(51), Wt, 64);
  launch_gemm_bias_relu_f32_64(stream, embBf, Wt, P(52), hF32, NS);
  // out = h @ out2.W + out2.b   [NS]
  rowdot64_bias_k<<<ceil_div(NS, 8), T, 0, stream>>>(hF32, P(53), P(54), (float*)d_out, NS);
}
